// GATLayer_45243185496592
// MI455X (gfx1250) — matrix operand order repaired and checked by reading, not compiled, not run on hardware
//
#include <hip/hip_runtime.h>
#include <hip/hip_bf16.h>

typedef __attribute__((ext_vector_type(16))) _Float16 v16h;
typedef __attribute__((ext_vector_type(8)))  float    v8f;

#define NN   4096
#define INF_ 128
#define HOF  128   // H*OUT_F

// ---- CDNA5 async memory->LDS helpers (ASYNCcnt-tracked, saddr form) -------
__device__ __forceinline__ unsigned lds_addr(const void* p) {
  return (unsigned)(unsigned long long)p;   // low 32 bits of generic = LDS offset
}
__device__ __forceinline__ void async_load_b32_s(unsigned lds, unsigned voff,
                                                 const void* sbase) {
  asm volatile("global_load_async_to_lds_b32 %0, %1, %2"
               :: "v"(lds), "v"(voff), "s"((unsigned long long)sbase) : "memory");
}
__device__ __forceinline__ void async_load_b128_s(unsigned lds, unsigned voff,
                                                  const void* sbase) {
  asm volatile("global_load_async_to_lds_b128 %0, %1, %2"
               :: "v"(lds), "v"(voff), "s"((unsigned long long)sbase) : "memory");
}
__device__ __forceinline__ void wait_async0() {
  asm volatile("s_wait_asynccnt 0x0" ::: "memory");
}

// ---------------------------------------------------------------------------
// Kernel 1: Wx = x @ W (fp32); store Wx^T f16 and the factorized softmax
// tables E1=exp(e_src), E2=exp(0.2*e_src), F1=exp(e_dst), F2=exp(0.2*e_dst).
// ---------------------------------------------------------------------------
__global__ __launch_bounds__(128) void gat_wx_kernel(
    const float* __restrict__ x, const float* __restrict__ W,
    const float* __restrict__ a,
    _Float16* __restrict__ wxT, float* __restrict__ E1g,
    float* __restrict__ E2g, float* __restrict__ F1g,
    float* __restrict__ F2g) {
  __shared__ float xrow[INF_];
  const int n = blockIdx.x;
  const int t = threadIdx.x;
  xrow[t] = x[(size_t)n * INF_ + t];
  __syncthreads();

  float acc = 0.f;
#pragma unroll 8
  for (int k = 0; k < INF_; ++k)
    acc = fmaf(xrow[k], W[k * HOF + t], acc);

  wxT[(size_t)t * NN + n] = (_Float16)acc;

  const int h = t >> 5;
  const int f = t & 31;
  float ps = acc * a[h * 64 + f];
  float pd = acc * a[h * 64 + 32 + f];
#pragma unroll
  for (int off = 16; off > 0; off >>= 1) {
    ps += __shfl_xor(ps, off, 32);
    pd += __shfl_xor(pd, off, 32);
  }
  if (f == 0) {
    E1g[n * 4 + h] = __expf(ps);
    E2g[n * 4 + h] = __expf(0.2f * ps);
    F1g[n * 4 + h] = __expf(pd);
    F2g[n * 4 + h] = __expf(0.2f * pd);
  }
}

// ---------------------------------------------------------------------------
// Kernel 2: fused masked-softmax attention + aggregation.
// Numerator: exp(lrelu(es+ed)) = max(E1*F1, E2*F2)  (exp monotone, slope<1)
// Mask: p &= -(adj)   (adj in {0,1} -> all-ones / zero bitmask)
// K=64 per iteration (2x v_wmma_f32_16x16x32_f16), double-buffered async
// memory->LDS staging.
// ---------------------------------------------------------------------------
__global__ __launch_bounds__(256) void gat_attn_kernel(
    const int* __restrict__ adj, const _Float16* __restrict__ wxT,
    const float* __restrict__ E1g, const float* __restrict__ E2g,
    const float* __restrict__ F1g, const float* __restrict__ F2g,
    float* __restrict__ out) {
  __shared__ __align__(16) int      adj_lds[2][16][68];   // 272B rows
  __shared__ __align__(16) _Float16 wx_lds[2][HOF][72];   // 144B rows
  __shared__ __align__(16) float    f1_lds[2][4][68];
  __shared__ __align__(16) float    f2_lds[2][4][68];
  __shared__ __align__(16) float    bc[8][16];            // per-wave broadcast

  const int i0    = blockIdx.x * 16;
  const int t     = threadIdx.x;
  const int wave  = t >> 5;
  const int lane  = t & 31;
  const int h     = wave >> 1;
  const int colb  = wave * 16;
  const int row   = lane & 15;
  const int kbase = (lane < 16) ? 0 : 8;   // A-fragment K base (ISA layout)
  const int koff  = (lane < 16) ? 0 : 16;  // unused; B fragment also uses kbase

  // --- per-thread staging assignments (loop-invariant byte offsets) ---
  const int ar = t >> 4, ac = (t & 15) * 4;          // adj: 16B per thread
  const int wcol = t >> 1, whb = (t & 1) * 32;       // wxT: 64B per thread
  const unsigned voA = ((unsigned)(i0 + ar) * NN + ac) * 4u;
  const unsigned voW = ((unsigned)wcol * NN + whb) * 2u;
  const unsigned voF = (unsigned)t * 4u;

  unsigned lA[2], lW[2], lF1[2], lF2[2];
#pragma unroll
  for (int b = 0; b < 2; ++b) {
    lA[b]  = lds_addr(&adj_lds[b][ar][ac]);
    lW[b]  = lds_addr(&wx_lds[b][wcol][whb]);
    lF1[b] = lds_addr(&f1_lds[b][t & 3][t >> 2]);
    lF2[b] = lds_addr(&f2_lds[b][t & 3][t >> 2]);
  }

  // --- per-row softmax factors ---
  const float E1 = E1g[(i0 + row) * 4 + h];
  const float E2 = E2g[(i0 + row) * 4 + h];

  float l = 0.f;
  v8f acc = {};

  // --- prologue: async-stage tile 0 into buffer 0 ---
  async_load_b128_s(lA[0], voA, adj);
#pragma unroll
  for (int q = 0; q < 4; ++q)
    async_load_b128_s(lW[0] + 16 * q, voW + 16 * q, wxT);
  async_load_b32_s(lF1[0], voF, F1g);
  async_load_b32_s(lF2[0], voF, F2g);

  for (int jt = 0; jt < NN; jt += 64) {
    const int cur = (jt >> 6) & 1;
    wait_async0();        // own tile-DMA writes done
    __syncthreads();      // everyone's writes visible; prior reads retired

    // --- kick off next tile's DMA into the other buffer (uniform SGPR base) --
    if (jt + 64 < NN) {
      const int nxt = cur ^ 1;
      const size_t j4 = (size_t)(jt + 64);
      async_load_b128_s(lA[nxt], voA, (const char*)adj + j4 * 4);
#pragma unroll
      for (int q = 0; q < 4; ++q)
        async_load_b128_s(lW[nxt] + 16 * q, voW + 16 * q,
                          (const char*)wxT + j4 * 2);
      async_load_b32_s(lF1[nxt], voF, (const char*)F1g + j4 * 16);
      async_load_b32_s(lF2[nxt], voF, (const char*)F2g + j4 * 16);
    }

    // --- masked softmax numerators for 2x32 columns, pack A fragments ---
    v16h pa[2];
    float psum = 0.f;
#pragma unroll
    for (int s = 0; s < 2; ++s) {
      const int ko = s * 32 + kbase;
      int4 av4[4];
      {
        const int4* p0 = (const int4*)&adj_lds[cur][row][ko];
        const int4* p1 = (const int4*)&adj_lds[cur][row][ko + 16];
        av4[0] = p0[0]; av4[1] = p0[1]; av4[2] = p1[0]; av4[3] = p1[1];
      }
      const int* av = (const int*)av4;
      float4 f1v4[4], f2v4[4];
      {
        const float4* q0 = (const float4*)&f1_lds[cur][h][ko];
        const float4* q1 = (const float4*)&f1_lds[cur][h][ko + 16];
        f1v4[0] = q0[0]; f1v4[1] = q0[1]; f1v4[2] = q1[0]; f1v4[3] = q1[1];
        const float4* r0 = (const float4*)&f2_lds[cur][h][ko];
        const float4* r1 = (const float4*)&f2_lds[cur][h][ko + 16];
        f2v4[0] = r0[0]; f2v4[1] = r0[1]; f2v4[2] = r1[0]; f2v4[3] = r1[1];
      }
      const float* f1v = (const float*)f1v4;
      const float* f2v = (const float*)f2v4;
#pragma unroll
      for (int e = 0; e < 16; ++e) {
        const float t1 = E1 * f1v[e];
        const float t2 = E2 * f2v[e];
        const float pm = fmaxf(t1, t2);            // exp(leaky_relu(es+ed))
        const unsigned msk = (unsigned)(-av[e]);   // adj mask: 0 or ~0
        const float p = __uint_as_float(__float_as_uint(pm) & msk);
        psum += p;
        pa[s][e] = (_Float16)p;
      }
    }
    psum += __shfl_xor(psum, 16, 32);
    l += psum;

    // --- B fragments from LDS, two K-steps of WMMA ---
    v16h pb0, pb1;
    {
      const uint4* bp0 = (const uint4*)&wx_lds[cur][colb + row][kbase];
      *(uint4*)&pb0       = bp0[0];
      *((uint4*)&pb0 + 1) = bp0[2];
      const uint4* bp1 = (const uint4*)&wx_lds[cur][colb + row][32 + kbase];
      *(uint4*)&pb1       = bp1[0];
      *((uint4*)&pb1 + 1) = bp1[2];
    }
    acc = __builtin_amdgcn_wmma_f32_16x16x32_f16(
        false, pa[0], false, pb0, (short)0, acc, false, false);
    acc = __builtin_amdgcn_wmma_f32_16x16x32_f16(
        false, pa[1], false, pb1, (short)0, acc, false, false);
  }

  // --- finalize: divide by softmax denominator, store 16x16 tile ---
  if (lane < 16) bc[wave][row] = l;
  {
    const float4* l4 = (const float4*)&bc[wave][(lane < 16) ? 0 : 8];
    const float4 la = l4[0], lb = l4[1];
    const float lfv[8] = {la.x, la.y, la.z, la.w, lb.x, lb.y, lb.z, lb.w};
#pragma unroll
    for (int r = 0; r < 8; ++r) {
      const int M = r + ((lane < 16) ? 0 : 8);
      out[(size_t)(i0 + M) * HOF + colb + row] = acc[r] / lfv[r];
    }
  }
}

// ---------------------------------------------------------------------------
extern "C" void kernel_launch(void* const* d_in, const int* in_sizes, int n_in,
                              void* d_out, int out_size, void* d_ws, size_t ws_size,
                              hipStream_t stream) {
  const float* x   = (const float*)d_in[0];   // (4096,128) f32
  const int*   adj = (const int*)d_in[1];     // (4096,4096) i32
  const float* W   = (const float*)d_in[2];   // (128,128) f32
  const float* a   = (const float*)d_in[3];   // (4,64) f32
  float* out = (float*)d_out;                 // (4096,128) f32

  char* ws = (char*)d_ws;
  _Float16* wxT = (_Float16*)ws;                          // 1 MB
  float* E1g    = (float*)(ws + (size_t)HOF * NN * 2);    // 64 KB
  float* E2g    = E1g + NN * 4;                           // 64 KB
  float* F1g    = E2g + NN * 4;                           // 64 KB
  float* F2g    = F1g + NN * 4;                           // 64 KB

  gat_wx_kernel<<<NN, 128, 0, stream>>>(x, W, a, wxT, E1g, E2g, F1g, F2g);
  gat_attn_kernel<<<NN / 16, 256, 0, stream>>>(adj, wxT, E1g, E2g, F1g, F2g, out);
}
